// MultiHeadAttention_12189117186065
// MI455X (gfx1250) — compile-verified, matrix operand order checked by reading, not run on hardware
//
#include <hip/hip_runtime.h>
#include <hip/hip_bf16.h>
#include <stdint.h>

typedef __attribute__((ext_vector_type(16))) __bf16 v16bf;
typedef __attribute__((ext_vector_type(8)))  float  v8f;

#define WMMA_BF16(A, B, C) \
  __builtin_amdgcn_wmma_f32_16x16x32_bf16(false, (A), false, (B), (short)0, (C), false, false)

// B=4, S=2048, E=1024, H=16, D=64; M = B*S = 8192
static constexpr int S_LEN = 2048;
static constexpr int EMB   = 1024;

// per-lane K index inside a 16x32 (16-bit) WMMA A/B fragment (ISA 7.12.2):
// k = (e&7) + 16*(e>>3) + 8*(lane>>4)
__device__ __forceinline__ int kmap(int e, int lane) {
  return (e & 7) | ((e >> 3) << 4) | ((lane & 16) >> 1);
}
// inverse: given k (0..31): e = (k&7)|((k>>4)<<3); lane_hi_bit = (k&8)<<1

// fragment-major tile = [32 lanes][16 bf16] = 1KB; each lane reads its 32B contiguously
__device__ __forceinline__ v16bf load_frag(const __bf16* tile, int lane) {
  union { int4 q[2]; v16bf v; } u;
  const int4* p = reinterpret_cast<const int4*>(tile) + lane * 2;
  u.q[0] = p[0];
  u.q[1] = p[1];
  return u.v;
}

__device__ __forceinline__ unsigned pack2(float x, float y) {
  union { __bf16 h[2]; unsigned u; } c;
  c.h[0] = (__bf16)x;
  c.h[1] = (__bf16)y;
  return c.u;
}

// CDNA5 async global->LDS copy (16B per lane), tracked by ASYNCcnt
typedef __attribute__((address_space(3))) char lds_char;
__device__ __forceinline__ void async_copy16(void* lds_dst, const void* gsrc) {
  unsigned loff = (unsigned)(unsigned long long)(lds_char*)lds_dst;
  asm volatile("global_load_async_to_lds_b128 %0, %1, off" ::"v"(loff), "v"(gsrc) : "memory");
}
__device__ __forceinline__ void wait_async() {
  asm volatile("s_wait_asynccnt 0x0" ::: "memory");
}

// ---------- activation convert + swizzle to A-fragment-major [mt][kc][lane][e] ----------
__global__ void swz_act(const float* __restrict__ src, unsigned* __restrict__ dst) {
  int idx = blockIdx.x * 256 + threadIdx.x;  // bf16-pair index; total 512*32*32*8
  int j = idx & 7;
  int lane = (idx >> 3) & 31;
  int kc = (idx >> 8) & 31;
  int mt = idx >> 13;  // 0..511
  int row = mt * 16 + (lane & 15);
  int col = kc * 32 + kmap(2 * j, lane);
  float2 f = *reinterpret_cast<const float2*>(src + (size_t)row * EMB + col);
  dst[idx] = pack2(f.x, f.y);
}

// ---------- weight convert + swizzle to B-fragment-major [nt][kc][lane][e] ----------
__global__ void swz_wgt(const float* __restrict__ W, unsigned* __restrict__ dst) {
  const int nt = blockIdx.x;  // 0..63
  const int kc = blockIdx.y;  // 0..31
  __shared__ float tile[32][17];
  const int t = threadIdx.x;
  {
    int r = t >> 3, c = (t & 7) * 2;  // 32 k-rows x 16 n-cols, coalesced
    float2 f = *reinterpret_cast<const float2*>(W + (size_t)(kc * 32 + r) * EMB + nt * 16 + c);
    tile[r][c] = f.x;
    tile[r][c + 1] = f.y;
  }
  __syncthreads();
  int j = t & 7, lane = t >> 3;
  int k0 = kmap(2 * j, lane);
  dst[(((size_t)nt * 32 + kc) * 32 + lane) * 8 + j] =
      pack2(tile[k0][lane & 15], tile[k0 + 1][lane & 15]);
}

// ---------- pack bool mask (1 byte/elem) into bitmask (1 bit/elem) ----------
__global__ void pack_mask(const uint8_t* __restrict__ m, unsigned* __restrict__ out) {
  int idx = blockIdx.x * 256 + threadIdx.x;  // word index; total 4*2048*2048/32 = 524288
  const uint4* p = reinterpret_cast<const uint4*>(m + (size_t)idx * 32);
  uint4 a = p[0], b = p[1];
  unsigned wsrc[8] = {a.x, a.y, a.z, a.w, b.x, b.y, b.z, b.w};
  unsigned r = 0;
#pragma unroll
  for (int wi = 0; wi < 8; ++wi)
#pragma unroll
    for (int j = 0; j < 4; ++j) r |= ((wsrc[wi] >> (j * 8)) & 1u) << (wi * 4 + j);
  out[idx] = r;
}

// ---------- QKV projection GEMM, LDS-free, fragments straight from swizzled global ----------
// block = 256x128 tile (16x8 16-tiles), 8 waves each 4(M)x4(N) WMMA tiles, K-chunks of 32
__launch_bounds__(256)
__global__ void gemm_qkv(const __bf16* __restrict__ Xq, const __bf16* __restrict__ Xk,
                         const __bf16* __restrict__ Xv,
                         const __bf16* __restrict__ Wq, const __bf16* __restrict__ Wk,
                         const __bf16* __restrict__ Wv,
                         const float* __restrict__ bq, const float* __restrict__ bk,
                         const float* __restrict__ bv,
                         __bf16* __restrict__ Qsw, __bf16* __restrict__ Ksw,
                         __bf16* __restrict__ Vsw) {
  const int z = blockIdx.z;
  const __bf16* X = (z == 0) ? Xq : (z == 1) ? Xk : Xv;
  const __bf16* W = (z == 0) ? Wq : (z == 1) ? Wk : Wv;
  const float* bias = (z == 0) ? bq : (z == 1) ? bk : bv;

  const int t = threadIdx.x, lane = t & 31, w = t >> 5;
  const int mt0 = blockIdx.x * 16 + (w & 3) * 4;  // wave's first m-tile
  const int nt0 = blockIdx.y * 8 + (w >> 2) * 4;  // wave's first n-tile

  v8f acc[4][4];
#pragma unroll
  for (int m = 0; m < 4; ++m)
#pragma unroll
    for (int n = 0; n < 4; ++n)
#pragma unroll
      for (int e = 0; e < 8; ++e) acc[m][n][e] = 0.f;

  for (int kc = 0; kc < 32; ++kc) {
    v16bf af[4], bfr[4];
#pragma unroll
    for (int m = 0; m < 4; ++m) af[m] = load_frag(X + ((size_t)(mt0 + m) * 32 + kc) * 512, lane);
#pragma unroll
    for (int n = 0; n < 4; ++n) bfr[n] = load_frag(W + ((size_t)(nt0 + n) * 32 + kc) * 512, lane);
    if (kc + 1 < 32) {
      __builtin_prefetch(X + ((size_t)mt0 * 32 + kc + 1) * 512 + lane * 16, 0, 3);
      __builtin_prefetch(W + ((size_t)nt0 * 32 + kc + 1) * 512 + lane * 16, 0, 3);
    }
#pragma unroll
    for (int m = 0; m < 4; ++m)
#pragma unroll
      for (int n = 0; n < 4; ++n) acc[m][n] = WMMA_BF16(af[m], bfr[n], acc[m][n]);
  }

  // epilogue: + bias, write directly in the layout flash-attention consumes
#pragma unroll
  for (int m = 0; m < 4; ++m)
#pragma unroll
    for (int n = 0; n < 4; ++n) {
      const int col = (nt0 + n) * 16 + (lane & 15);
      const float bval = bias[col];
#pragma unroll
      for (int i = 0; i < 8; ++i) {
        const int row = (mt0 + m) * 16 + i + ((lane & 16) >> 1);
        const float val = acc[m][n][i] + bval;
        if (z < 2) {
          // Q/K: [rowtile 512][h 16][dc 2][lane 32][e 16]; value at (row, d=dc*32+k5)
          int h = col >> 6, d = col & 63, dc = (d >> 5) & 1, k5 = d & 31;
          int e = (k5 & 7) | ((k5 >> 4) << 3);
          int lane2 = (row & 15) | ((k5 & 8) << 1);
          __bf16* dst = (z == 0) ? Qsw : Ksw;
          dst[((((size_t)(row >> 4) * 16 + h) * 2 + dc) * 32 + lane2) * 16 + e] = (__bf16)val;
        } else {
          // V: [b 4][h 16][dt 4][kck 64][lane 32][e 16]; B-matrix k=key, n=d
          int h = col >> 6, d = col & 63, dt = (d >> 4) & 3;
          int s = row & (S_LEN - 1), kck = s >> 5, k5 = s & 31;
          int e = (k5 & 7) | ((k5 >> 4) << 3);
          int lane2 = (d & 15) | ((k5 & 8) << 1);
          Vsw[(((((size_t)(row >> 11) * 16 + h) * 4 + dt) * 64 + kck) * 32 + lane2) * 16 + e] =
              (__bf16)val;
        }
      }
    }
}

// ---------- flash attention ----------
__launch_bounds__(256)
__global__ void flash_attn(const __bf16* __restrict__ Qsw, const __bf16* __restrict__ Ksw,
                           const __bf16* __restrict__ Vsw, const uint4* __restrict__ maskbits,
                           __bf16* __restrict__ Osw) {
  const int t = threadIdx.x, lane = t & 31, w = t >> 5;
  const int qt = blockIdx.x * 128;
  const int b = blockIdx.y >> 4, h = blockIdx.y & 15;

  __shared__ __bf16 Kls[8 * 2 * 512];  // [ktile][dc][lane][e]  16KB
  __shared__ __bf16 Vls[4 * 4 * 512];  // [dt][kck][lane][e]    16KB
  __shared__ __bf16 Pls[8][4 * 512];   // per-wave [kk][lane][e] 32KB

  // Q fragments for this wave's 16 rows (held all kernel)
  v16bf qf[2];
  {
    size_t qt16 = (size_t)(b * S_LEN + qt) / 16 + w;
#pragma unroll
    for (int dc = 0; dc < 2; ++dc) qf[dc] = load_frag(Qsw + ((qt16 * 16 + h) * 2 + dc) * 512, lane);
  }

  v8f oacc[4];
#pragma unroll
  for (int n = 0; n < 4; ++n)
#pragma unroll
    for (int e = 0; e < 8; ++e) oacc[n][e] = 0.f;
  float mrow[8], lrow[8];
#pragma unroll
  for (int i = 0; i < 8; ++i) { mrow[i] = -3.0e38f; lrow[i] = 0.f; }

  for (int kc = 0; kc < S_LEN; kc += 128) {
    __syncthreads();
    const size_t kt0 = (size_t)(b * S_LEN + kc) >> 4;
    // async-copy K chunk (8 key-tiles x 2 d-chunks, contiguous 2KB per key-tile)
#pragma unroll
    for (int i = 0; i < 4; ++i) {
      int uid = t + i * 256;  // 1024 x 16B
      int ktl = uid >> 7, off = uid & 127;
      const __bf16* src = Ksw + ((kt0 + ktl) * 16 + h) * 2 * 512 + off * 8;
      async_copy16(&Kls[uid * 8], src);
    }
    // async-copy V chunk (4 d-tiles x 4 key-chunks of 32)
#pragma unroll
    for (int i = 0; i < 4; ++i) {
      int uid = t + i * 256;
      int dt = uid >> 8, off = uid & 255;
      const __bf16* src =
          Vsw + ((((size_t)(b * 16 + h) * 4 + dt) * 64 + (kc >> 5)) * 512) + off * 8;
      async_copy16(&Vls[uid * 8], src);
    }
    wait_async();
    __syncthreads();

    // scores S = Q K^T
    v8f sc[8];
#pragma unroll
    for (int nt = 0; nt < 8; ++nt) {
      v8f c;
#pragma unroll
      for (int e = 0; e < 8; ++e) c[e] = 0.f;
#pragma unroll
      for (int dc = 0; dc < 2; ++dc) {
        v16bf bfrag = load_frag(&Kls[(nt * 2 + dc) * 512], lane);
        c = WMMA_BF16(qf[dc], bfrag, c);
      }
      sc[nt] = c;
    }

    // scale + mask via packed bitmask: one 128-bit segment per owned row
#pragma unroll
    for (int i = 0; i < 8; ++i) {
      int row = qt + w * 16 + i + ((lane & 16) >> 1);
      uint4 m4 = maskbits[((size_t)(b * S_LEN + row) << 4) + (kc >> 7)];
      unsigned mw[4] = {m4.x, m4.y, m4.z, m4.w};
#pragma unroll
      for (int nt = 0; nt < 8; ++nt) {
        unsigned bit = (mw[nt >> 1] >> (((nt & 1) << 4) + (lane & 15))) & 1u;
        float sv = sc[nt][i] * 0.125f;  // 1/sqrt(64)
        sc[nt][i] = bit ? -1.0e30f : sv;
      }
    }

    // online softmax (rows live in 16-lane halves)
#pragma unroll
    for (int i = 0; i < 8; ++i) {
      float rmax = -3.0e38f;
#pragma unroll
      for (int nt = 0; nt < 8; ++nt) rmax = fmaxf(rmax, sc[nt][i]);
#pragma unroll
      for (int off = 8; off >= 1; off >>= 1) rmax = fmaxf(rmax, __shfl_xor(rmax, off, 32));
      float mnew = fmaxf(mrow[i], rmax);
      float corr = __expf(mrow[i] - mnew);
      mrow[i] = mnew;
      float psum = 0.f;
#pragma unroll
      for (int nt = 0; nt < 8; ++nt) {
        float p = __expf(sc[nt][i] - mnew);
        sc[nt][i] = p;
        psum += p;
      }
#pragma unroll
      for (int off = 8; off >= 1; off >>= 1) psum += __shfl_xor(psum, off, 32);
      lrow[i] = lrow[i] * corr + psum;
#pragma unroll
      for (int nt2 = 0; nt2 < 4; ++nt2) oacc[nt2][i] *= corr;
    }

    // scatter P into its A-fragment layout (per-wave private LDS region)
#pragma unroll
    for (int nt = 0; nt < 8; ++nt) {
      int kk = nt >> 1;
      int e = (lane & 7) | ((nt & 1) << 3);
      int hb = (lane & 8) << 1;
#pragma unroll
      for (int i = 0; i < 8; ++i) {
        int r = i + ((lane & 16) >> 1);
        Pls[w][kk * 512 + (r | hb) * 16 + e] = (__bf16)sc[nt][i];
      }
    }

    // O += P V
#pragma unroll
    for (int kk = 0; kk < 4; ++kk) {
      v16bf afrag = load_frag(&Pls[w][kk * 512], lane);
#pragma unroll
      for (int nt2 = 0; nt2 < 4; ++nt2) {
        v16bf bfrag = load_frag(&Vls[(nt2 * 4 + kk) * 512], lane);
        oacc[nt2] = WMMA_BF16(afrag, bfrag, oacc[nt2]);
      }
    }
  }

  // normalize; write in A-fragment layout [rowtile 512][kc 32][lane][e] for gemm_out
#pragma unroll
  for (int nt2 = 0; nt2 < 4; ++nt2) {
    int col = h * 64 + nt2 * 16 + (lane & 15);
    int kc_o = col >> 5;
    int e = (lane & 7) | ((nt2 & 1) << 3);
    int hb = (lane & 8) << 1;
#pragma unroll
    for (int i = 0; i < 8; ++i) {
      int R = b * S_LEN + qt + w * 16 + i + ((lane & 16) >> 1);
      Osw[(((size_t)(R >> 4) * 32 + kc_o) * 32 + ((R & 15) | hb)) * 16 + e] =
          (__bf16)(oacc[nt2][i] / lrow[i]);
    }
  }
}

// ---------- output projection + bias + residual -> fp32 linear ----------
__launch_bounds__(256)
__global__ void gemm_out(const __bf16* __restrict__ X, const __bf16* __restrict__ W,
                         const float* __restrict__ bias, const float* __restrict__ resid,
                         float* __restrict__ O) {
  const int t = threadIdx.x, lane = t & 31, w = t >> 5;
  const int mt0 = blockIdx.x * 16 + (w & 3) * 4;
  const int nt0 = blockIdx.y * 8 + (w >> 2) * 4;

  v8f acc[4][4];
#pragma unroll
  for (int m = 0; m < 4; ++m)
#pragma unroll
    for (int n = 0; n < 4; ++n)
#pragma unroll
      for (int e = 0; e < 8; ++e) acc[m][n][e] = 0.f;

  for (int kc = 0; kc < 32; ++kc) {
    v16bf af[4], bfr[4];
#pragma unroll
    for (int m = 0; m < 4; ++m) af[m] = load_frag(X + ((size_t)(mt0 + m) * 32 + kc) * 512, lane);
#pragma unroll
    for (int n = 0; n < 4; ++n) bfr[n] = load_frag(W + ((size_t)(nt0 + n) * 32 + kc) * 512, lane);
    if (kc + 1 < 32) {
      __builtin_prefetch(X + ((size_t)mt0 * 32 + kc + 1) * 512 + lane * 16, 0, 3);
      __builtin_prefetch(W + ((size_t)nt0 * 32 + kc + 1) * 512 + lane * 16, 0, 3);
    }
#pragma unroll
    for (int m = 0; m < 4; ++m)
#pragma unroll
      for (int n = 0; n < 4; ++n) acc[m][n] = WMMA_BF16(af[m], bfr[n], acc[m][n]);
  }

#pragma unroll
  for (int m = 0; m < 4; ++m)
#pragma unroll
    for (int n = 0; n < 4; ++n) {
      const int col = (nt0 + n) * 16 + (lane & 15);
      const float bval = bias[col];
#pragma unroll
      for (int i = 0; i < 8; ++i) {
        const int row = (mt0 + m) * 16 + i + ((lane & 16) >> 1);
        O[(size_t)row * EMB + col] = acc[m][n][i] + bval + resid[(size_t)row * EMB + col];
      }
    }
}

// ---------- LayerNorm over last dim (1024), one block per row ----------
__launch_bounds__(256)
__global__ void layernorm_kernel(const float* __restrict__ x, const float* __restrict__ gamma,
                                 const float* __restrict__ beta, float* __restrict__ out) {
  const int row = blockIdx.x;
  const int t = threadIdx.x;
  const float* xr = x + (size_t)row * EMB;
  float v[4], s = 0.f, sq = 0.f;
#pragma unroll
  for (int i = 0; i < 4; ++i) {
    v[i] = xr[t + i * 256];
    s += v[i];
    sq += v[i] * v[i];
  }
#pragma unroll
  for (int off = 16; off >= 1; off >>= 1) {
    s += __shfl_xor(s, off, 32);
    sq += __shfl_xor(sq, off, 32);
  }
  __shared__ float rs[8], rq[8];
  if ((t & 31) == 0) { rs[t >> 5] = s; rq[t >> 5] = sq; }
  __syncthreads();
  float ts = 0.f, tq = 0.f;
#pragma unroll
  for (int i = 0; i < 8; ++i) { ts += rs[i]; tq += rq[i]; }
  float mu = ts * (1.f / EMB);
  float var = tq * (1.f / EMB) - mu * mu;
  float inv = rsqrtf(var + 1e-5f);
#pragma unroll
  for (int i = 0; i < 4; ++i) {
    int c = t + i * 256;
    out[(size_t)row * EMB + c] = (v[i] - mu) * inv * gamma[c] + beta[c];
  }
}

extern "C" void kernel_launch(void* const* d_in, const int* in_sizes, int n_in,
                              void* d_out, int out_size, void* d_ws, size_t ws_size,
                              hipStream_t stream) {
  const float* q = (const float*)d_in[0];
  const float* k = (const float*)d_in[1];
  const float* v = (const float*)d_in[2];
  const uint8_t* mask = (const uint8_t*)d_in[3];
  const float* Wq = (const float*)d_in[4];
  const float* bq = (const float*)d_in[5];
  const float* Wk = (const float*)d_in[6];
  const float* bk = (const float*)d_in[7];
  const float* Wv = (const float*)d_in[8];
  const float* bv = (const float*)d_in[9];
  const float* Wf = (const float*)d_in[10];
  const float* bf = (const float*)d_in[11];
  const float* gamma = (const float*)d_in[12];
  const float* beta = (const float*)d_in[13];
  float* out = (float*)d_out;

  const size_t MN = (size_t)8192 * 1024;
  const size_t WW = (size_t)1024 * 1024;

  char* p = (char*)d_ws;
  auto alloc = [&](size_t bytes) {
    void* r = (void*)p;
    p += (bytes + 255) & ~(size_t)255;
    return r;
  };
  __bf16* Xqsw = (__bf16*)alloc(MN * 2);
  __bf16* Xksw = (__bf16*)alloc(MN * 2);
  __bf16* Xvsw = (__bf16*)alloc(MN * 2);
  __bf16* Wqb = (__bf16*)alloc(WW * 2);
  __bf16* Wkb = (__bf16*)alloc(WW * 2);
  __bf16* Wvb = (__bf16*)alloc(WW * 2);
  __bf16* Wfb = (__bf16*)alloc(WW * 2);
  __bf16* Qsw = (__bf16*)alloc(MN * 2);
  __bf16* Ksw = (__bf16*)alloc(MN * 2);
  __bf16* Vsw = (__bf16*)alloc(MN * 2);
  unsigned* mbits = (unsigned*)alloc((size_t)4 * 2048 * 2048 / 8);
  __bf16* Osw = Xqsw;        // attn out aliases dead Xqsw
  float* Xf = (float*)Xksw;  // pre-LN fp32 aliases dead Xksw+Xvsw

  // convert + swizzle activations and weights; pack mask to bits
  swz_act<<<16384, 256, 0, stream>>>(q, (unsigned*)Xqsw);
  swz_act<<<16384, 256, 0, stream>>>(k, (unsigned*)Xksw);
  swz_act<<<16384, 256, 0, stream>>>(v, (unsigned*)Xvsw);
  swz_wgt<<<dim3(64, 32), 256, 0, stream>>>(Wq, (unsigned*)Wqb);
  swz_wgt<<<dim3(64, 32), 256, 0, stream>>>(Wk, (unsigned*)Wkb);
  swz_wgt<<<dim3(64, 32), 256, 0, stream>>>(Wv, (unsigned*)Wvb);
  swz_wgt<<<dim3(64, 32), 256, 0, stream>>>(Wf, (unsigned*)Wfb);
  pack_mask<<<2048, 256, 0, stream>>>(mask, mbits);

  // projections -> swizzled Q/K/V
  gemm_qkv<<<dim3(32, 8, 3), 256, 0, stream>>>(Xqsw, Xksw, Xvsw, Wqb, Wkb, Wvb, bq, bk, bv, Qsw,
                                               Ksw, Vsw);

  // attention
  flash_attn<<<dim3(16, 64), 256, 0, stream>>>(Qsw, Ksw, Vsw, (const uint4*)mbits, Osw);

  // output projection + residual
  gemm_out<<<dim3(32, 8), 256, 0, stream>>>(Osw, Wfb, bf, q, Xf);

  // LayerNorm
  layernorm_kernel<<<8192, 256, 0, stream>>>(Xf, gamma, beta, out);
}
